// SocialPool_70703751627229
// MI455X (gfx1250) — matrix operand order repaired and checked by reading, not compiled, not run on hardware
//
#include <hip/hip_runtime.h>
#include <math.h>

typedef __attribute__((ext_vector_type(16))) _Float16 v16h;
typedef __attribute__((ext_vector_type(8)))  _Float16 v8h;
typedef __attribute__((ext_vector_type(8)))  float    v8f;

#define NAG   4096
#define HID   48
#define FCIN  1728
#define FCOUT 48

// ring: ceil( (5/log(8)) * log(r/0.5) ) = ceil( (5/(2*log(8))) * log(4*r^2) )
#define RING_SCALE_HALF 1.2022458674074695f
#define SQRT3 1.7320508075688772f

__device__ __forceinline__ v8f wmma16x16x32(v16h a, v16h b, v8f c) {
  // D = A(16x32 f16) x B(32x16 f16) + C(f32)
  return __builtin_amdgcn_wmma_f32_16x16x32_f16(
      false, a, false, b, (short)0, c, false, false);
}

// ---------------------------------------------------------------------------
// Kernel 0: convert hidden -> f16, transpose W[48][1728] -> Wt16[1728][48]
// ---------------------------------------------------------------------------
__global__ void prep_kernel(const float* __restrict__ hidden,
                            const float* __restrict__ W,
                            _Float16* __restrict__ hidden16,
                            _Float16* __restrict__ wt16) {
  int stride = gridDim.x * blockDim.x;
  int t = blockIdx.x * blockDim.x + threadIdx.x;
  for (int p = t; p < NAG * HID; p += stride)
    hidden16[p] = (_Float16)hidden[p];
  for (int p = t; p < FCIN * FCOUT; p += stride) {
    int k = p / FCOUT, o = p - k * FCOUT;
    wt16[p] = (_Float16)W[(size_t)o * FCIN + k];   // Wt[k][o] = W[o][k]
  }
}

// ---------------------------------------------------------------------------
// Kernel 1: per-agent pooling via one-hot WMMA. One wave (32 threads) = one
// agent. Accumulators: acc[bin_tile 0..2][n_tile 0..2 = hid, 3 = counts].
// ---------------------------------------------------------------------------
__global__ __launch_bounds__(32) void pool_kernel(
    const float* __restrict__ y,            // [NAG][2]
    const _Float16* __restrict__ hidden16,  // [NAG][HID]
    _Float16* __restrict__ pooled16)        // [NAG][FCIN]
{
  __shared__ __align__(16) int smidx[32];
  const int i      = blockIdx.x;
  const int lane   = threadIdx.x;      // 0..31
  const int half16 = lane & 16;        // 0 or 16
  const int base1  = half16 >> 1;      // 0 or 8 (A-slot K base / upper-M offset)
  const int m      = lane & 15;

  const float yi_x = y[2 * i + 0];
  const float yi_y = y[2 * i + 1];

  v8f acc[3][4];
#pragma unroll
  for (int bt = 0; bt < 3; ++bt)
#pragma unroll
    for (int nt = 0; nt < 4; ++nt)
      acc[bt][nt] = {};

  // Constant ones-column B: B[k][0] = 1 for every k -> D[:,0] = bin counts.
  v16h bones = {};
  for (int s = 0; s < 16; ++s) bones[s] = (m == 0) ? (_Float16)1.0f : (_Float16)0.0f;

  for (int j0 = 0; j0 < NAG; j0 += 32) {
    const int jj = j0 + lane;

    // --- issue all 3 hidden B-tile loads up front (separate regs -> one
    //     clause, latency hidden under the idx/one-hot VALU below) ---
    v16h B0, B1, B2;
#pragma unroll
    for (int s = 0; s < 16; ++s) {
      const int kb = j0 + base1 + (s < 8 ? s : 8 + s);
      const _Float16* hrow = hidden16 + (size_t)kb * HID + m;
      B0[s] = hrow[0]; B1[s] = hrow[16]; B2[s] = hrow[32];
    }

    // --- log-polar bin index for pair (i, jj) ---
    float dx = y[2 * jj + 0] - yi_x;
    float dy = y[2 * jj + 1] - yi_y;
    float r2 = dx * dx + dy * dy;
    // ring = ceil(2.40449 * log(2*r)) ; log(2r) = 0.5*log(4*r^2), no sqrt.
    float ringf = ceilf(RING_SCALE_HALF * __logf(4.0f * r2));
    if (!(ringf >= -1.0e30f)) ringf = 0.0f;              // -inf -> 0
    int ring = (int)ringf;
    // wedge = sextant of atan2(dy,dx): boundaries at multiples of pi/3,
    // decided by sign tests against the +-60 degree lines.
    float c1 = dy - SQRT3 * dx;   // >0  <=> theta >  pi/3   (upper half)
    float c2 = dy + SQRT3 * dx;   // <0  <=> theta > 2pi/3 (upper) ; >=0 <=> theta >= -pi/3 (lower)
    int wedge;
    if (dy >= 0.0f)
      wedge = (c1 <= 0.0f) ? 2 : ((c2 >= 0.0f) ? 3 : 4);
    else
      wedge = (c2 >= 0.0f) ? 1 : 0;
    int idxv = ring * 6 + wedge;
    if (idxv < 6 || idxv >= 36) idxv = 0;

    smidx[lane] = idxv;
    __syncthreads();   // single-wave WG: barrier -> S_NOP, keeps DS ordering

    // A-fragment K slots: s=0..7 -> K=base1+s ; s=8..15 -> K=base1+16+(s-8)
    int kidx[16];
    {
      const int4* p = (const int4*)smidx;
      int b4 = base1 >> 2;              // 0 or 2
      int4 q0 = p[b4 + 0];
      int4 q1 = p[b4 + 1];
      int4 q2 = p[b4 + 4];
      int4 q3 = p[b4 + 5];
      kidx[0] = q0.x;  kidx[1] = q0.y;  kidx[2]  = q0.z;  kidx[3]  = q0.w;
      kidx[4] = q1.x;  kidx[5] = q1.y;  kidx[6]  = q1.z;  kidx[7]  = q1.w;
      kidx[8] = q2.x;  kidx[9] = q2.y;  kidx[10] = q2.z;  kidx[11] = q2.w;
      kidx[12] = q3.x; kidx[13] = q3.y; kidx[14] = q3.z;  kidx[15] = q3.w;
    }
    __syncthreads();

    // One-hot A per bin tile (row M = m, bins bt*16+m), exact 0/1 in f16.
    v16h A[3];
#pragma unroll
    for (int bt = 0; bt < 3; ++bt) {
      const int target = bt * 16 + m;
      v16h a;
#pragma unroll
      for (int s = 0; s < 16; ++s)
        a[s] = (kidx[s] == target) ? (_Float16)1.0f : (_Float16)0.0f;
      A[bt] = a;
    }


#pragma unroll
    for (int bt = 0; bt < 3; ++bt) {
      acc[bt][0] = wmma16x16x32(A[bt], B0, acc[bt][0]);
      acc[bt][1] = wmma16x16x32(A[bt], B1, acc[bt][1]);
      acc[bt][2] = wmma16x16x32(A[bt], B2, acc[bt][2]);
      acc[bt][3] = wmma16x16x32(A[bt], bones, acc[bt][3]);
    }
  }

  // mean = sums / max(cnt,1); write bins 6..41 as features 0..35.
  // C/D layout: lane, reg r -> (M = r + base1, N = m). Counts at N==0 live in
  // lanes 0 (M=0..7) and 16 (M=8..15): broadcast within each half-wave.
#pragma unroll
  for (int bt = 0; bt < 3; ++bt) {
    float cnt[8];
#pragma unroll
    for (int r = 0; r < 8; ++r)
      cnt[r] = __shfl(acc[bt][3][r], half16, 32);
#pragma unroll
    for (int r = 0; r < 8; ++r) {
      const int bin = bt * 16 + r + base1;
      const float inv = __builtin_amdgcn_rcpf(fmaxf(cnt[r], 1.0f));
      if (bin >= 6 && bin < 42) {
        const int fb = bin - 6;
#pragma unroll
        for (int nt = 0; nt < 3; ++nt) {
          // bins 36..41 are unreachable by idx -> force exact zeros
          float mval = (bin < 36) ? acc[bt][nt][r] * inv : 0.0f;
          pooled16[(size_t)i * FCIN + fb * HID + nt * 16 + m] = (_Float16)mval;
        }
      }
    }
  }
}

// ---------------------------------------------------------------------------
// Kernel 2: out = relu(pooled @ Wt + b), WMMA over K=1728. One wave per
// 16x16 output tile: 256 row tiles x 3 col tiles = 768 waves.
// ---------------------------------------------------------------------------
__global__ __launch_bounds__(32) void fc_kernel(
    const _Float16* __restrict__ pooled16,  // [NAG][FCIN]
    const _Float16* __restrict__ wt16,      // [FCIN][FCOUT]
    const float* __restrict__ bias,         // [FCOUT]
    float* __restrict__ out)                // [NAG][FCOUT]
{
  const int tile = blockIdx.x;
  const int rt = tile / 3, ct = tile - 3 * rt;
  const int lane  = threadIdx.x;
  const int m     = lane & 15;
  const int base1 = (lane & 16) >> 1;      // 0 or 8
  const size_t arow = (size_t)(rt * 16 + m) * FCIN;   // A row = agent

  v8f acc = {};
  for (int kk = 0; kk < FCIN; kk += 32) {
    const v8h* ap = (const v8h*)(pooled16 + arow + kk + base1);
    v8h alo = ap[0];
    v8h ahi = ap[2];   // +16 halfs
    v16h A = __builtin_shufflevector(alo, ahi, 0, 1, 2, 3, 4, 5, 6, 7,
                                     8, 9, 10, 11, 12, 13, 14, 15);
    v16h B;
#pragma unroll
    for (int s = 0; s < 16; ++s) {
      const int kb = kk + base1 + (s < 8 ? s : 8 + s);
      B[s] = wt16[(size_t)kb * FCOUT + ct * 16 + m];
    }
    acc = wmma16x16x32(A, B, acc);
  }

  const float bv = bias[ct * 16 + m];      // N is fixed per lane
#pragma unroll
  for (int r = 0; r < 8; ++r) {
    const int row = rt * 16 + r + base1;
    float v = fmaxf(acc[r] + bv, 0.0f);
    out[(size_t)row * FCOUT + ct * 16 + m] = v;
  }
}

// ---------------------------------------------------------------------------
extern "C" void kernel_launch(void* const* d_in, const int* in_sizes, int n_in,
                              void* d_out, int out_size, void* d_ws, size_t ws_size,
                              hipStream_t stream) {
  const float* y_pred = (const float*)d_in[0];
  // d_in[1] = x_start (unused by the reference)
  const float* hidden = (const float*)d_in[2];
  const float* W      = (const float*)d_in[3];
  const float* b      = (const float*)d_in[4];
  float* out = (float*)d_out;

  char* ws = (char*)d_ws;
  _Float16* hidden16 = (_Float16*)(ws);                         // 393,216 B
  _Float16* wt16     = (_Float16*)(ws + 393216);                // 165,888 B
  _Float16* pooled16 = (_Float16*)(ws + 393216 + 165888);       // 14,155,776 B
  (void)in_sizes; (void)n_in; (void)out_size; (void)ws_size;

  prep_kernel<<<512, 256, 0, stream>>>(hidden, W, hidden16, wt16);
  pool_kernel<<<NAG, 32, 0, stream>>>(y_pred, hidden16, pooled16);
  fc_kernel<<<(NAG / 16) * 3, 32, 0, stream>>>(pooled16, wt16, b, out);
}
